// CollabDeterminer_7773890806136
// MI455X (gfx1250) — compile-verified, matrix operand order checked by reading, not run on hardware
//
#include <hip/hip_runtime.h>
#include <math.h>

#define IN_DIM   384
#define HID      64
#define N_COLLAB 64
#define N_CTX    100000

#define LOG_VAR2f (-4.605170185988091f)   // log(0.01)
#define INV_VAR2f (100.0f)                // 1/0.01

typedef __attribute__((ext_vector_type(16))) __bf16 v16bf;
typedef __attribute__((ext_vector_type(8)))  __bf16 v8bf;
typedef __attribute__((ext_vector_type(4)))  __bf16 v4bf;
typedef __attribute__((ext_vector_type(8)))  float  v8f;

// ---- workspace layout (offsets in bf16 halfs) ----
// B-fragment buffers: each fragment = 32 lanes * 16 halfs = 512 halfs.
#define OFF_W1   0        // 4 ntiles * 12 kchunks = 48 frags
#define OFF_W21  24576    // 8 frags
#define OFF_W22  28672    // 8 frags
#define OFF_W3   32768    // 8 frags
#define OFF_W4   36864    // 24 ntiles * 2 kchunks = 48 frags
#define OFF_CZ   61440    // 8 frags (collab_z^T)
#define OFF_END_HALFS 65536
#define OFF_COLLABZ_BYTES (OFF_END_HALFS * 2)   // 64x64 f32 normalized collab z

static __device__ inline v16bf bf_cat(v8bf lo, v8bf hi) {
  v16bf r;
#pragma unroll
  for (int i = 0; i < 8; ++i) { r[i] = lo[i]; r[i + 8] = hi[i]; }
  return r;
}

// A-fragment (16-bit, 16x32) from an LDS row (row-major, element K contiguous).
// lane: M = lane&15, half = lane>>4. elem i -> K = kBase + (i&7) + half*8 + (i>>3)*16
static __device__ inline v16bf loadA(const __bf16* rowp, int kBase, int hf) {
  const __bf16* p = rowp + kBase + hf * 8;
  return bf_cat(*(const v8bf*)p, *(const v8bf*)(p + 16));
}

// B-fragment from a pre-packed fragment buffer: 16 contiguous halfs per lane.
static __device__ inline v16bf loadB(const __bf16* fragbuf, int fragIdx, int lane) {
  const __bf16* p = fragbuf + ((size_t)fragIdx * 32 + lane) * 16;
  return bf_cat(*(const v8bf*)p, *(const v8bf*)(p + 8));
}

static __device__ inline v8f wmma_bf16(v16bf a, v16bf b, v8f c) {
  return __builtin_amdgcn_wmma_f32_16x16x32_bf16(false, a, false, b, (short)0, c,
                                                 false, false);
}

// -------------------- kernel 0: zero the loss slot --------------------
__global__ void zero_loss_kernel(float* loss) {
  if (threadIdx.x == 0) *loss = 0.0f;
}

// -------------------- kernel 1: collab VAE (64 rows, exact f32) --------------------
__global__ void __launch_bounds__(64) collab_kernel(
    const float* __restrict__ collabs, const float* __restrict__ epsc,
    const float* __restrict__ W1,  const float* __restrict__ b1,
    const float* __restrict__ W21, const float* __restrict__ b21,
    const float* __restrict__ W22, const float* __restrict__ b22,
    const float* __restrict__ W3,  const float* __restrict__ b3,
    const float* __restrict__ W4,  const float* __restrict__ b4,
    float* __restrict__ collabZ, float* __restrict__ outLoss) {
  __shared__ float hsh[HID], zsh[HID], h2sh[HID], red[HID];
  const int j = threadIdx.x;
  const int row = blockIdx.x;
  const float* x = collabs + (size_t)row * IN_DIM;

  float s = b1[j];
  for (int k = 0; k < IN_DIM; ++k) s += x[k] * W1[j * IN_DIM + k];
  hsh[j] = fmaxf(s, 0.0f);
  __syncthreads();

  float mu = b21[j], lv = b22[j];
  for (int k = 0; k < HID; ++k) {
    mu += hsh[k] * W21[j * HID + k];
    lv += hsh[k] * W22[j * HID + k];
  }
  const float z = mu + epsc[(size_t)row * HID + j] * __expf(0.5f * lv) * 0.1f;
  const float kterm = 1.0f - LOG_VAR2f + lv - (mu * mu + __expf(lv)) * INV_VAR2f;
  zsh[j] = z;
  red[j] = z * z;
  __syncthreads();
  for (int o = 32; o > 0; o >>= 1) {
    if (j < o) red[j] += red[j + o];
    __syncthreads();
  }
  const float norm = fmaxf(sqrtf(red[0]), 1e-12f);
  __syncthreads();
  collabZ[(size_t)row * HID + j] = z / norm;

  float h2 = b3[j];
  for (int k = 0; k < HID; ++k) h2 += zsh[k] * W3[j * HID + k];
  h2sh[j] = fmaxf(h2, 0.0f);
  __syncthreads();

  float macc = 0.0f;
  for (int seg = 0; seg < IN_DIM / HID; ++seg) {
    const int col = j + HID * seg;
    float xh = b4[col];
    for (int k = 0; k < HID; ++k) xh += h2sh[k] * W4[col * HID + k];
    const float d = xh - x[col];
    macc += d * d;
  }
  red[j] = macc * (1.0f / ((float)N_COLLAB * IN_DIM)) +
           kterm * (-0.5f / ((float)N_COLLAB * HID));
  __syncthreads();
  for (int o = 32; o > 0; o >>= 1) {
    if (j < o) red[j] += red[j + o];
    __syncthreads();
  }
  if (j == 0) atomicAdd(outLoss, red[0]);
}

// -------------------- kernel 2: pack collab_z^T into B-fragments --------------------
__global__ void __launch_bounds__(32) pack_cz_kernel(const float* __restrict__ collabZ,
                                                     __bf16* __restrict__ fragCZ) {
  const int f = blockIdx.x;          // 8 frags: nt*2 + kc
  const int lane = threadIdx.x;
  const int nt = f >> 1, kc = f & 1;
  const int hf = lane >> 4, rr = lane & 15;
  const int n = nt * 16 + rr;
  __bf16* dst = fragCZ + ((size_t)f * 32 + lane) * 16;
#pragma unroll
  for (int i = 0; i < 16; ++i) {
    const int k = kc * 32 + (i & 7) + hf * 8 + ((i >> 3) << 4);
    dst[i] = (__bf16)collabZ[(size_t)n * HID + k];  // B[k][n] = z[n][k]
  }
}

// -------------------- kernel 3: pack context-VAE weights into B-fragments ----------
__global__ void __launch_bounds__(32) pack_weights_kernel(
    const float* __restrict__ W1, const float* __restrict__ W21,
    const float* __restrict__ W22, const float* __restrict__ W3,
    const float* __restrict__ W4, __bf16* __restrict__ wsb) {
  const int f = blockIdx.x;  // 0..119
  const int lane = threadIdx.x;
  const int hf = lane >> 4, rr = lane & 15;
  const float* W;
  __bf16* dst;
  int nt, kc, K;
  if (f < 48) {            // fc1: [64][384], 4 ntiles x 12 kchunks
    W = W1; K = IN_DIM; nt = f / 12; kc = f % 12;
    dst = wsb + OFF_W1 + ((size_t)f * 32 + lane) * 16;
  } else if (f < 56) {     // fc21: [64][64]
    const int fi = f - 48; W = W21; K = HID; nt = fi >> 1; kc = fi & 1;
    dst = wsb + OFF_W21 + ((size_t)fi * 32 + lane) * 16;
  } else if (f < 64) {     // fc22
    const int fi = f - 56; W = W22; K = HID; nt = fi >> 1; kc = fi & 1;
    dst = wsb + OFF_W22 + ((size_t)fi * 32 + lane) * 16;
  } else if (f < 72) {     // fc3
    const int fi = f - 64; W = W3; K = HID; nt = fi >> 1; kc = fi & 1;
    dst = wsb + OFF_W3 + ((size_t)fi * 32 + lane) * 16;
  } else {                 // fc4: [384][64], 24 ntiles x 2 kchunks
    const int fi = f - 72; W = W4; K = HID; nt = fi >> 1; kc = fi & 1;
    dst = wsb + OFF_W4 + ((size_t)fi * 32 + lane) * 16;
  }
  const int n = nt * 16 + rr;
#pragma unroll
  for (int i = 0; i < 16; ++i) {
    const int k = kc * 32 + (i & 7) + hf * 8 + ((i >> 3) << 4);
    dst[i] = (__bf16)W[(size_t)n * K + k];  // B[k][n] = W[n][k]
  }
}

// -------------------- kernel 4: fused context VAE + scores + sampling --------------
__global__ void __launch_bounds__(32) context_kernel(
    const float* __restrict__ ctx, const float* __restrict__ eps,
    const float* __restrict__ rnd,
    const float* __restrict__ b1, const float* __restrict__ b21,
    const float* __restrict__ b22, const float* __restrict__ b3,
    const float* __restrict__ b4,
    const __bf16* __restrict__ fragW1, const __bf16* __restrict__ fragW21,
    const __bf16* __restrict__ fragW22, const __bf16* __restrict__ fragW3,
    const __bf16* __restrict__ fragW4, const __bf16* __restrict__ fragCZ,
    const float* __restrict__ collabZ,
    float* __restrict__ outIdx, float* __restrict__ outLogp,
    float* __restrict__ outEmb, float* __restrict__ outLoss) {
  __shared__ __align__(16) __bf16 Xs[16][IN_DIM];
  __shared__ __align__(16) __bf16 Hs[16][HID];
  __shared__ __align__(16) __bf16 Zs[16][HID];
  __shared__ __align__(16) __bf16 ZN[16][HID];
  __shared__ __align__(16) __bf16 H2s[16][HID];
  __shared__ float Zf[16][HID];
  __shared__ float Ss[16][HID];
  __shared__ float normS[16];
  __shared__ int selS[16];

  const int lane = threadIdx.x;
  const int hf = lane >> 4;   // half of the wave
  const int rr = lane & 15;   // row (A/M) or col (C/N) index
  const int rowBase = blockIdx.x * 16;

  // ---- cooperative load of the 16x384 context tile -> bf16 LDS (float4 global loads)
#pragma unroll 4
  for (int t = 0; t < 48; ++t) {
    const int idx = lane + 32 * t;          // 0..1535 over 16 rows * 96 float4
    const int row = idx / 96, q = idx % 96;
    const float4 v = ((const float4*)(ctx + (size_t)(rowBase + row) * IN_DIM))[q];
    v4bf b;
    b[0] = (__bf16)v.x; b[1] = (__bf16)v.y; b[2] = (__bf16)v.z; b[3] = (__bf16)v.w;
    *(v4bf*)&Xs[row][q * 4] = b;
  }

  // ---- fc1: H = relu(X @ W1^T + b1)   (K=384 -> 12 chunks, N=64 -> 4 tiles)
  v8f acc[4];
#pragma unroll
  for (int nt = 0; nt < 4; ++nt) {
    const float bb = b1[nt * 16 + rr];
#pragma unroll
    for (int r = 0; r < 8; ++r) acc[nt][r] = bb;
  }
#pragma unroll
  for (int kc = 0; kc < 12; ++kc) {
    const v16bf a = loadA(&Xs[rr][0], kc * 32, hf);
#pragma unroll
    for (int nt = 0; nt < 4; ++nt)
      acc[nt] = wmma_bf16(a, loadB(fragW1, nt * 12 + kc, lane), acc[nt]);
  }
#pragma unroll
  for (int nt = 0; nt < 4; ++nt)
#pragma unroll
    for (int r = 0; r < 8; ++r)
      Hs[r + 8 * hf][nt * 16 + rr] = (__bf16)fmaxf(acc[nt][r], 0.0f);

  // ---- fc21 / fc22: mu, logvar   (K=64 -> 2 chunks)
  v8f mu[4], lv[4];
#pragma unroll
  for (int nt = 0; nt < 4; ++nt) {
    const float bm = b21[nt * 16 + rr], bl = b22[nt * 16 + rr];
#pragma unroll
    for (int r = 0; r < 8; ++r) { mu[nt][r] = bm; lv[nt][r] = bl; }
  }
#pragma unroll
  for (int kc = 0; kc < 2; ++kc) {
    const v16bf a = loadA(&Hs[rr][0], kc * 32, hf);
#pragma unroll
    for (int nt = 0; nt < 4; ++nt) {
      mu[nt] = wmma_bf16(a, loadB(fragW21, nt * 2 + kc, lane), mu[nt]);
      lv[nt] = wmma_bf16(a, loadB(fragW22, nt * 2 + kc, lane), lv[nt]);
    }
  }

  // ---- reparameterize + KLD partial
  float kacc = 0.0f;
#pragma unroll
  for (int nt = 0; nt < 4; ++nt)
#pragma unroll
    for (int r = 0; r < 8; ++r) {
      const int row = r + 8 * hf, col = nt * 16 + rr;
      const float m = mu[nt][r], l = lv[nt][r];
      const float e = eps[(size_t)(rowBase + row) * HID + col];
      const float z = m + e * __expf(0.5f * l) * 0.1f;
      Zs[row][col] = (__bf16)z;
      Zf[row][col] = z;
      kacc += 1.0f - LOG_VAR2f + l - (m * m + __expf(l)) * INV_VAR2f;
    }

  // ---- per-row L2 norms (lanes 0..15, one row each)
  if (lane < 16) {
    float s = 0.0f;
    for (int k = 0; k < HID; ++k) s += Zf[lane][k] * Zf[lane][k];
    normS[lane] = fmaxf(sqrtf(s), 1e-12f);
  }
#pragma unroll
  for (int nt = 0; nt < 4; ++nt)
#pragma unroll
    for (int r = 0; r < 8; ++r) {
      const int row = r + 8 * hf, col = nt * 16 + rr;
      ZN[row][col] = (__bf16)(Zf[row][col] / normS[row]);
    }

  // ---- fc3: H2 = relu(z @ W3^T + b3)
  v8f a3[4];
#pragma unroll
  for (int nt = 0; nt < 4; ++nt) {
    const float bb = b3[nt * 16 + rr];
#pragma unroll
    for (int r = 0; r < 8; ++r) a3[nt][r] = bb;
  }
#pragma unroll
  for (int kc = 0; kc < 2; ++kc) {
    const v16bf a = loadA(&Zs[rr][0], kc * 32, hf);
#pragma unroll
    for (int nt = 0; nt < 4; ++nt)
      a3[nt] = wmma_bf16(a, loadB(fragW3, nt * 2 + kc, lane), a3[nt]);
  }
#pragma unroll
  for (int nt = 0; nt < 4; ++nt)
#pragma unroll
    for (int r = 0; r < 8; ++r)
      H2s[r + 8 * hf][nt * 16 + rr] = (__bf16)fmaxf(a3[nt][r], 0.0f);

  // ---- fc4 + fused MSE   (N=384 -> 24 tiles, K=64 -> 2 chunks)
  float macc = 0.0f;
  {
    const v16bf a0 = loadA(&H2s[rr][0], 0, hf);
    const v16bf a1 = loadA(&H2s[rr][0], 32, hf);
#pragma unroll 4
    for (int nt = 0; nt < 24; ++nt) {
      v8f c;
      const float bb = b4[nt * 16 + rr];
#pragma unroll
      for (int r = 0; r < 8; ++r) c[r] = bb;
      c = wmma_bf16(a0, loadB(fragW4, nt * 2 + 0, lane), c);
      c = wmma_bf16(a1, loadB(fragW4, nt * 2 + 1, lane), c);
#pragma unroll
      for (int r = 0; r < 8; ++r) {
        const int row = r + 8 * hf, col = nt * 16 + rr;
        const float d = c[r] - (float)Xs[row][col];
        macc += d * d;
      }
    }
  }

  // ---- scores: S = zn @ collab_z^T   (K=64, N=64)
  {
    const v16bf a0 = loadA(&ZN[rr][0], 0, hf);
    const v16bf a1 = loadA(&ZN[rr][0], 32, hf);
#pragma unroll
    for (int nt = 0; nt < 4; ++nt) {
      v8f c;
#pragma unroll
      for (int r = 0; r < 8; ++r) c[r] = 0.0f;
      c = wmma_bf16(a0, loadB(fragCZ, nt * 2 + 0, lane), c);
      c = wmma_bf16(a1, loadB(fragCZ, nt * 2 + 1, lane), c);
#pragma unroll
      for (int r = 0; r < 8; ++r) Ss[r + 8 * hf][nt * 16 + rr] = c[r];
    }
  }

  // ---- softmax + cumsum sampling + log-prob (lanes 0..15, one row each)
  if (lane < 16) {
    const int row = lane;
    float m = Ss[row][0];
    for (int k = 1; k < HID; ++k) m = fmaxf(m, Ss[row][k]);
    float s = 0.0f;
    for (int k = 0; k < HID; ++k) s += __expf(Ss[row][k] - m);
    const float inv = 1.0f / s;
    const float r = rnd[rowBase + row];
    int sel = -1;
    float cum = 0.0f;
    for (int k = 0; k < HID; ++k) {
      cum += __expf(Ss[row][k] - m) * inv;
      if (cum > r && sel < 0) sel = k;
    }
    if (sel < 0) sel = 0;  // argmax of all-False -> 0, matching jnp.argmax
    selS[row] = sel;
    outIdx[rowBase + row] = (float)sel;
    outLogp[rowBase + row] = (Ss[row][sel] - m) - __logf(s);
  }

  // ---- gather collab embedding (f32 normalized collab z from workspace)
  for (int row = 0; row < 16; ++row) {
    const int sel = selS[row];
#pragma unroll
    for (int cc = 0; cc < 2; ++cc) {
      const int col = lane + 32 * cc;
      outEmb[(size_t)(rowBase + row) * HID + col] = collabZ[(size_t)sel * HID + col];
    }
  }

  // ---- loss partials: wave reduce + one atomic
  float tot = macc * (1.0f / ((float)N_CTX * IN_DIM)) +
              kacc * (-0.5f / ((float)N_CTX * HID));
  for (int o = 16; o > 0; o >>= 1) tot += __shfl_down(tot, o, 32);
  if (lane == 0) atomicAdd(outLoss, tot);
}

// -------------------------------- launcher --------------------------------
extern "C" void kernel_launch(void* const* d_in, const int* in_sizes, int n_in,
                              void* d_out, int out_size, void* d_ws, size_t ws_size,
                              hipStream_t stream) {
  const float* collabs     = (const float*)d_in[0];
  const float* contexts    = (const float*)d_in[1];
  const float* eps_collab  = (const float*)d_in[2];
  const float* eps_context = (const float*)d_in[3];
  const float* random_num  = (const float*)d_in[4];
  // c_params leaves (dict order): fc1_w, fc1_b, fc21_w, fc21_b, fc22_w, fc22_b,
  //                               fc3_w, fc3_b, fc4_w, fc4_b
  const float* cW1  = (const float*)d_in[5];
  const float* cb1  = (const float*)d_in[6];
  const float* cW21 = (const float*)d_in[7];
  const float* cb21 = (const float*)d_in[8];
  const float* cW22 = (const float*)d_in[9];
  const float* cb22 = (const float*)d_in[10];
  const float* cW3  = (const float*)d_in[11];
  const float* cb3  = (const float*)d_in[12];
  const float* cW4  = (const float*)d_in[13];
  const float* cb4  = (const float*)d_in[14];
  // x_params leaves
  const float* xW1  = (const float*)d_in[15];
  const float* xb1  = (const float*)d_in[16];
  const float* xW21 = (const float*)d_in[17];
  const float* xb21 = (const float*)d_in[18];
  const float* xW22 = (const float*)d_in[19];
  const float* xb22 = (const float*)d_in[20];
  const float* xW3  = (const float*)d_in[21];
  const float* xb3  = (const float*)d_in[22];
  const float* xW4  = (const float*)d_in[23];
  const float* xb4  = (const float*)d_in[24];

  __bf16* wsb = (__bf16*)d_ws;
  __bf16* fragW1  = wsb + OFF_W1;
  __bf16* fragW21 = wsb + OFF_W21;
  __bf16* fragW22 = wsb + OFF_W22;
  __bf16* fragW3  = wsb + OFF_W3;
  __bf16* fragW4  = wsb + OFF_W4;
  __bf16* fragCZ  = wsb + OFF_CZ;
  float* collabZ  = (float*)((char*)d_ws + OFF_COLLABZ_BYTES);

  float* out     = (float*)d_out;
  float* outIdx  = out;
  float* outLogp = out + N_CTX;
  float* outEmb  = out + 2 * N_CTX;
  float* outLoss = out + 2 * N_CTX + (size_t)N_CTX * HID;

  zero_loss_kernel<<<1, 32, 0, stream>>>(outLoss);
  collab_kernel<<<N_COLLAB, 64, 0, stream>>>(collabs, eps_collab, cW1, cb1, cW21,
                                             cb21, cW22, cb22, cW3, cb3, cW4, cb4,
                                             collabZ, outLoss);
  pack_cz_kernel<<<8, 32, 0, stream>>>(collabZ, fragCZ);
  pack_weights_kernel<<<120, 32, 0, stream>>>(xW1, xW21, xW22, xW3, xW4, wsb);
  context_kernel<<<N_CTX / 16, 32, 0, stream>>>(
      contexts, eps_context, random_num, xb1, xb21, xb22, xb3, xb4, fragW1,
      fragW21, fragW22, fragW3, fragW4, fragCZ, collabZ, outIdx, outLogp, outEmb,
      outLoss);
}
